// AttentionGatedMamba_83451214561942
// MI455X (gfx1250) — matrix operand order repaired and checked by reading, not compiled, not run on hardware
//
#include <hip/hip_runtime.h>
#include <hip/hip_bf16.h>
#include <stdint.h>

// ---------------- problem constants (match reference) ----------------
#define D_MODEL 512
#define D_STATE 16
#define D_CONV  4
#define D_INNER 1024          // 2 * D_MODEL
#define DT_RANK 32
#define BATCH   2
#define SEQ     2048
#define MROWS   (BATCH * SEQ) // 4096

typedef __attribute__((ext_vector_type(16))) __bf16 v16bf;
typedef __attribute__((ext_vector_type(8)))  float  v8f;

// ---------------- f32 -> bf16 convert ----------------
__global__ __launch_bounds__(256) void cvt_bf16_kernel(const float* __restrict__ src,
                                                       __hip_bfloat16* __restrict__ dst, int n) {
    int i = blockIdx.x * 256 + threadIdx.x;
    if (i < n) dst[i] = __float2bfloat16(src[i]);
}

// ---------------- generic bf16 WMMA GEMM: C[M,N] = A[M,K] * W[N,K]^T ----------------
// Register-blocked: each wave computes a 32x64 output region (2x4 subtiles of 16x16),
// so each k-step issues 8 v_wmma per 6 fragment loads (2 A-frags reused across 4 B-frags).
// EPI: 0 = plain store, 1 = +bias, 2 = +bias then out = gateSrc * sigmoid(acc)
template <int EPI>
__global__ __launch_bounds__(128)
void gemm_bf16_wmma(const __hip_bfloat16* __restrict__ A,
                    const __hip_bfloat16* __restrict__ W,
                    float* __restrict__ C,
                    const float* __restrict__ bias,
                    const float* __restrict__ gateSrc,
                    int M, int N, int K, int totalTiles)
{
    const int wave = threadIdx.x >> 5;
    const int lane = threadIdx.x & 31;
    const int tile = blockIdx.x * 4 + wave;
    if (tile >= totalTiles) return;           // uniform per-wave: EXEC stays all-1 for WMMA

    const int tilesN = N >> 6;                // number of 64-wide N groups
    const int tM = tile / tilesN;             // 32-row M group
    const int tN = tile - tM * tilesN;        // 64-col N group
    const int hi = lane >> 4;                 // lane group 0/1
    const int lr = lane & 15;

    const __hip_bfloat16* aRow[2];
#pragma unroll
    for (int i = 0; i < 2; ++i)
        aRow[i] = A + (size_t)(tM * 32 + i * 16 + lr) * (size_t)K;
    const __hip_bfloat16* wRow[4];
#pragma unroll
    for (int j = 0; j < 4; ++j)
        wRow[j] = W + (size_t)(tN * 64 + j * 16 + lr) * (size_t)K;

    v8f acc[2][4] = {};
    for (int k = 0; k < K; k += 32) {
        union { uint4 q[2]; v16bf v; } af[2], bfm[4];
        // A fragment (16x32 bf16): per-lane halves K = 8*hi + [0..7] and 8*hi + 16 + [0..7]
#pragma unroll
        for (int i = 0; i < 2; ++i) {
            af[i].q[0] = *(const uint4*)(aRow[i] + k + 8 * hi);
            af[i].q[1] = *(const uint4*)(aRow[i] + k + 8 * hi + 16);
        }
        // B fragment (32x16 bf16): per-lane halves K = 8*hi + [0..7] and 8*hi + 16 + [0..7]
#pragma unroll
        for (int j = 0; j < 4; ++j) {
            bfm[j].q[0] = *(const uint4*)(wRow[j] + k + 8 * hi);
            bfm[j].q[1] = *(const uint4*)(wRow[j] + k + 8 * hi + 16);
        }
#pragma unroll
        for (int i = 0; i < 2; ++i)
#pragma unroll
            for (int j = 0; j < 4; ++j)
                acc[i][j] = __builtin_amdgcn_wmma_f32_16x16x32_bf16(
                    false, af[i].v, false, bfm[j].v, (short)0, acc[i][j], false, false);
    }

#pragma unroll
    for (int i = 0; i < 2; ++i) {
#pragma unroll
        for (int j = 0; j < 4; ++j) {
            const int col = tN * 64 + j * 16 + lr;
            float bv = (EPI >= 1) ? bias[col] : 0.f;
#pragma unroll
            for (int r = 0; r < 8; ++r) {
                int row = tM * 32 + i * 16 + r + 8 * hi;
                float v = acc[i][j][r] + bv;
                if (EPI == 2) {
                    float mv = gateSrc[(size_t)row * N + col];
                    v = mv * (1.f / (1.f + __expf(-v)));
                }
                C[(size_t)row * N + col] = v;
            }
        }
    }
}

// ---------------- causal depthwise conv (k=4) + SiLU ----------------
__global__ __launch_bounds__(256)
void conv_silu_kernel(const float* __restrict__ xz,   // [MROWS, 2*D_INNER], xi = cols 0..1023
                      const float* __restrict__ cw,   // [D_INNER, D_CONV]
                      const float* __restrict__ cb,   // [D_INNER]
                      float* __restrict__ xc,         // [MROWS, D_INNER] f32
                      __hip_bfloat16* __restrict__ xcBf)
{
    int idx = blockIdx.x * 256 + threadIdx.x;         // MROWS * D_INNER threads
    if (idx >= MROWS * D_INNER) return;
    int e = idx & (D_INNER - 1);
    int m = idx >> 10;
    int l = m & (SEQ - 1);
    int b = m >> 11;
    float s = cb[e];
#pragma unroll
    for (int k = 0; k < D_CONV; ++k) {
        int li = l + k - (D_CONV - 1);
        if (li >= 0)
            s += xz[((size_t)(b * SEQ + li)) * (2 * D_INNER) + e] * cw[e * D_CONV + k];
    }
    float v = s / (1.f + __expf(-s));                  // SiLU
    xc[idx]   = v;
    xcBf[idx] = __float2bfloat16(v);
}

// ---------------- extract dt_raw cols [0..31] of dbc as bf16 [MROWS, 32] ----------------
__global__ __launch_bounds__(256)
void dt_extract_kernel(const float* __restrict__ dbc, __hip_bfloat16* __restrict__ dtRaw) {
    int idx = blockIdx.x * 256 + threadIdx.x;          // MROWS * 32
    if (idx >= MROWS * DT_RANK) return;
    int m = idx >> 5, r = idx & 31;
    dtRaw[idx] = __float2bfloat16(dbc[(size_t)m * (DT_RANK + 2 * D_STATE) + r]);
}

// ---------------- sequential selective-scan, fused softplus/dA/skip/gate ----------------
__global__ __launch_bounds__(256)
void scan_kernel(const float* __restrict__ dtLin,   // [MROWS, D_INNER] pre-softplus
                 const float* __restrict__ xc,      // [MROWS, D_INNER]
                 const float* __restrict__ dbc,     // [MROWS, 64]: dt|B|C
                 const float* __restrict__ xz,      // [MROWS, 2048] (z = cols 1024..)
                 const float* __restrict__ A_log,   // [D_INNER, D_STATE]
                 const float* __restrict__ Dp,      // [D_INNER]
                 __hip_bfloat16* __restrict__ yBf)  // [MROWS, D_INNER] gated output
{
    int t = blockIdx.x * 256 + threadIdx.x;            // BATCH * D_INNER = 2048 threads
    if (t >= BATCH * D_INNER) return;
    int e = t & (D_INNER - 1);
    int b = t >> 10;
    float a[D_STATE], h[D_STATE];
#pragma unroll
    for (int s = 0; s < D_STATE; ++s) { a[s] = -__expf(A_log[e * D_STATE + s]); h[s] = 0.f; }
    float De = Dp[e];
    for (int l = 0; l < SEQ; ++l) {
        size_t m = (size_t)b * SEQ + l;
        float dtv = dtLin[m * D_INNER + e];
        dtv = (dtv > 20.f) ? dtv : log1pf(__expf(dtv));        // softplus
        float xcv = xc[m * D_INNER + e];
        const float* bc = dbc + m * (DT_RANK + 2 * D_STATE);
        float y = 0.f;
#pragma unroll
        for (int s = 0; s < D_STATE; ++s) {
            float dA = __expf(dtv * a[s]);
            h[s] = dA * h[s] + dtv * bc[DT_RANK + s] * xcv;
            y += h[s] * bc[DT_RANK + D_STATE + s];
        }
        float zv  = xz[m * (2 * D_INNER) + D_INNER + e];
        float out = (y + xcv * De) * (zv / (1.f + __expf(-zv)));  // + D*x, * silu(z)
        yBf[m * D_INNER + e] = __float2bfloat16(out);
    }
}

// ---------------- pack [m | context] as bf16 [MROWS, 2*D_MODEL] ----------------
__global__ __launch_bounds__(256)
void pack_gin_kernel(const float* __restrict__ mF, const float* __restrict__ ctx,
                     __hip_bfloat16* __restrict__ gIn) {
    int idx = blockIdx.x * 256 + threadIdx.x;          // MROWS * 1024
    if (idx >= MROWS * 2 * D_MODEL) return;
    int m = idx >> 10, c = idx & 1023;
    float v = (c < D_MODEL) ? mF[(size_t)m * D_MODEL + c]
                            : ctx[(size_t)m * D_MODEL + (c - D_MODEL)];
    gIn[idx] = __float2bfloat16(v);
}

// ---------------- host-side orchestration ----------------
static inline size_t align256(size_t x) { return (x + 255) & ~(size_t)255; }

extern "C" void kernel_launch(void* const* d_in, const int* in_sizes, int n_in,
                              void* d_out, int out_size, void* d_ws, size_t ws_size,
                              hipStream_t stream) {
    const float* x        = (const float*)d_in[0];
    const float* context  = (const float*)d_in[1];
    const float* in_proj  = (const float*)d_in[2];
    const float* conv_w   = (const float*)d_in[3];
    const float* conv_b   = (const float*)d_in[4];
    const float* x_proj   = (const float*)d_in[5];
    const float* dt_proj  = (const float*)d_in[6];
    const float* dt_projb = (const float*)d_in[7];
    const float* A_log    = (const float*)d_in[8];
    const float* Dp       = (const float*)d_in[9];
    const float* out_proj = (const float*)d_in[10];
    const float* gate_w   = (const float*)d_in[11];
    const float* gate_b   = (const float*)d_in[12];
    float* out = (float*)d_out;

    char* ws = (char*)d_ws; size_t off = 0;
    auto take = [&](size_t bytes) { char* p = ws + off; off = align256(off + bytes); return p; };

    __hip_bfloat16* xBf     = (__hip_bfloat16*)take((size_t)MROWS * D_MODEL * 2);
    __hip_bfloat16* wInBf   = (__hip_bfloat16*)take((size_t)2 * D_INNER * D_MODEL * 2);
    float*          xz      = (float*)         take((size_t)MROWS * 2 * D_INNER * 4);
    float*          xc      = (float*)         take((size_t)MROWS * D_INNER * 4);
    __hip_bfloat16* xcBf    = (__hip_bfloat16*)take((size_t)MROWS * D_INNER * 2);
    __hip_bfloat16* wXpBf   = (__hip_bfloat16*)take((size_t)(DT_RANK + 2 * D_STATE) * D_INNER * 2);
    float*          dbc     = (float*)         take((size_t)MROWS * (DT_RANK + 2 * D_STATE) * 4);
    __hip_bfloat16* dtRawBf = (__hip_bfloat16*)take((size_t)MROWS * DT_RANK * 2);
    __hip_bfloat16* wDtBf   = (__hip_bfloat16*)take((size_t)D_INNER * DT_RANK * 2);
    float*          dtLin   = (float*)         take((size_t)MROWS * D_INNER * 4);
    __hip_bfloat16* yBf     = (__hip_bfloat16*)take((size_t)MROWS * D_INNER * 2);
    __hip_bfloat16* wOutBf  = (__hip_bfloat16*)take((size_t)D_MODEL * D_INNER * 2);
    float*          mF      = (float*)         take((size_t)MROWS * D_MODEL * 4);
    __hip_bfloat16* gInBf   = (__hip_bfloat16*)take((size_t)MROWS * 2 * D_MODEL * 2);
    __hip_bfloat16* wGateBf = (__hip_bfloat16*)take((size_t)D_MODEL * 2 * D_MODEL * 2);
    (void)ws_size; (void)n_in; (void)in_sizes; (void)out_size;

    auto cvt = [&](const float* s, __hip_bfloat16* d, int n) {
        cvt_bf16_kernel<<<(n + 255) / 256, 256, 0, stream>>>(s, d, n);
    };

    // 1. bf16 conversions of x + weights
    cvt(x,        xBf,     MROWS * D_MODEL);
    cvt(in_proj,  wInBf,   2 * D_INNER * D_MODEL);
    cvt(x_proj,   wXpBf,   (DT_RANK + 2 * D_STATE) * D_INNER);
    cvt(dt_proj,  wDtBf,   D_INNER * DT_RANK);
    cvt(out_proj, wOutBf,  D_MODEL * D_INNER);
    cvt(gate_w,   wGateBf, D_MODEL * 2 * D_MODEL);

    // 2. in_proj: xz[4096,2048] = x @ in_proj^T
    {
        int M = MROWS, N = 2 * D_INNER, K = D_MODEL;
        int tiles = (M / 32) * (N / 64);
        gemm_bf16_wmma<0><<<tiles / 4, 128, 0, stream>>>(xBf, wInBf, xz, nullptr, nullptr, M, N, K, tiles);
    }
    // 3. depthwise conv + SiLU
    conv_silu_kernel<<<(MROWS * D_INNER) / 256, 256, 0, stream>>>(xz, conv_w, conv_b, xc, xcBf);
    // 4. x_proj: dbc[4096,64] = xc @ x_proj^T
    {
        int M = MROWS, N = DT_RANK + 2 * D_STATE, K = D_INNER;
        int tiles = (M / 32) * (N / 64);
        gemm_bf16_wmma<0><<<tiles / 4, 128, 0, stream>>>(xcBf, wXpBf, dbc, nullptr, nullptr, M, N, K, tiles);
    }
    // 5. dt_proj: dtLin[4096,1024] = dt_raw @ dt_proj^T + dt_proj_b
    dt_extract_kernel<<<(MROWS * DT_RANK) / 256, 256, 0, stream>>>(dbc, dtRawBf);
    {
        int M = MROWS, N = D_INNER, K = DT_RANK;
        int tiles = (M / 32) * (N / 64);
        gemm_bf16_wmma<1><<<tiles / 4, 128, 0, stream>>>(dtRawBf, wDtBf, dtLin, dt_projb, nullptr, M, N, K, tiles);
    }
    // 6. selective scan (fused softplus + dA + skip + silu(z) gate) -> yBf
    scan_kernel<<<(BATCH * D_INNER) / 256, 256, 0, stream>>>(dtLin, xc, dbc, xz, A_log, Dp, yBf);
    // 7. out_proj: mF[4096,512] = y @ out_proj^T
    {
        int M = MROWS, N = D_MODEL, K = D_INNER;
        int tiles = (M / 32) * (N / 64);
        gemm_bf16_wmma<0><<<tiles / 4, 128, 0, stream>>>(yBf, wOutBf, mF, nullptr, nullptr, M, N, K, tiles);
    }
    // 8. concat [m | context] -> bf16
    pack_gin_kernel<<<(MROWS * 2 * D_MODEL) / 256, 256, 0, stream>>>(mF, context, gInBf);
    // 9. gate GEMM with fused sigmoid-gate epilogue -> d_out
    {
        int M = MROWS, N = D_MODEL, K = 2 * D_MODEL;
        int tiles = (M / 32) * (N / 64);
        gemm_bf16_wmma<2><<<tiles / 4, 128, 0, stream>>>(gInBf, wGateBf, out, gate_b, mF, M, N, K, tiles);
    }
}
